// DGCNN_58978490909135
// MI455X (gfx1250) — matrix operand order repaired and checked by reading, not compiled, not run on hardware
//
#include <hip/hip_runtime.h>
#include <math.h>

// ---------------------------------------------------------------------------
// DGCNN forward for MI455X (gfx1250, wave32).
//  - SpMM (segment_sum) via f32 atomics (data is L2-resident: <8MB).
//  - Dense GNN GEMMs via V_WMMA_F32_16X16X4_F32 (exact fp32).
//  - 13x13 conv (1->64ch) as implicit GEMM via V_WMMA_F32_16X16X32_F16,
//    fused with bias + AdaptiveMaxPool(30,97) so the 381MB pre-pool tensor
//    never reaches HBM (post-fusion traffic ~12MB total).
// ---------------------------------------------------------------------------

#define NN     15360      // total nodes (8 * 1920)
#define NPER   1920
#define NB     8
#define NEDGE  491520     // 15360 * 32
#define FEAT   128
#define LATENT 97
#define KPOOL  30
#define POOLH  64         // 1920 / 30
#define OC     64
#define KSZ    169        // 13*13
#define KPAD   192        // pad K to 6 * 32

typedef __attribute__((ext_vector_type(2)))  float    v2f;
typedef __attribute__((ext_vector_type(8)))  float    v8f;
typedef __attribute__((ext_vector_type(16))) _Float16 v16h;

// ---------------------------- tiny utility kernels -------------------------

__global__ void fill_f32(float* __restrict__ p, float v, int n) {
    int t = blockIdx.x * blockDim.x + threadIdx.x;
    if (t < n) p[t] = v;
}

__global__ void copy_f32(float* __restrict__ d, const float* __restrict__ s, int n) {
    int t = blockIdx.x * blockDim.x + threadIdx.x;
    if (t < n) d[t] = s[t];
}

__global__ void deg_scatter(const int* __restrict__ dst, float* __restrict__ deg, int ne) {
    int t = blockIdx.x * blockDim.x + threadIdx.x;
    if (t < ne) atomicAdd(&deg[dst[t]], 1.0f);
}

// pooled[dst[e]][f] += x[src[e]][f]   (F is a power of two; lf = log2(F))
__global__ void edge_scatter(const float* __restrict__ x,
                             const int* __restrict__ src,
                             const int* __restrict__ dst,
                             float* __restrict__ pooled,
                             int ne, int lf) {
    const int F = 1 << lf;
    int t = blockIdx.x * blockDim.x + threadIdx.x;
    if (t >= (ne << lf)) return;
    int e = t >> lf;
    int f = t & (F - 1);
    atomicAdd(&pooled[(size_t)dst[e] * F + f], x[(size_t)src[e] * F + f]);
}

// ------------------- GNN dense layer: x = tanh((A@W + b)/deg) --------------
// A: [NN, Fi] fp32, W: [Fi, 32], out 32 cols -> xnext [NN,32] and convbuf cols
// grid = NN/16 blocks, 64 threads (2 waves; wave w owns N-tile w*16).
// Exact fp32 math through V_WMMA_F32_16X16X4_F32.

__global__ void gnn_gemm32_wmma(const float* __restrict__ A, int Fi,
                                const float* __restrict__ W,
                                const float* __restrict__ bias,
                                const float* __restrict__ deg,
                                float* __restrict__ xnext,
                                float* __restrict__ convbuf, int col_off) {
    const int lane  = threadIdx.x & 31;
    const int wave  = threadIdx.x >> 5;     // 0..1
    const int m     = lane & 15;
    const int half  = lane >> 4;            // 0: lanes 0-15, 1: lanes 16-31
    const int mbase = blockIdx.x * 16;
    const int nbase = wave * 16;

    v8f c = {};
    for (int k = 0; k < Fi; k += 4) {
        // A 16x4 fragment: lanes0-15 -> K=k,k+1 ; lanes16-31 -> K=k+2,k+3
        const float* ap = A + (size_t)(mbase + m) * Fi + k + 2 * half;
        v2f a; a.x = ap[0]; a.y = ap[1];
        // B 4x16 fragment: VGPR0 rows K=k / k+2, VGPR1 rows K=k+1 / k+3
        const float* bp = W + (size_t)(k + 2 * half) * 32 + nbase + m;
        v2f b; b.x = bp[0]; b.y = bp[32];
        c = __builtin_amdgcn_wmma_f32_16x16x4_f32(false, a, false, b,
                                                  (short)0, c, false, false);
    }

    const int   col = nbase + m;
    const float bb  = bias[col];
#pragma unroll
    for (int r = 0; r < 8; ++r) {
        const int row = mbase + r + 8 * half;        // C/D: M = r (+8 hi half)
        float v = tanhf((c[r] + bb) / deg[row]);
        xnext[(size_t)row * 32 + col] = v;
        convbuf[(size_t)row * LATENT + col_off + col] = v;
    }
}

// --------------------------- last GNN layer (Fo = 1) -----------------------

__global__ void gnn_layer3(const float* __restrict__ pooled,
                           const float* __restrict__ W3,
                           const float* __restrict__ b3,
                           const float* __restrict__ deg,
                           float* __restrict__ convbuf) {
    int n = blockIdx.x * blockDim.x + threadIdx.x;
    if (n >= NN) return;
    float s = b3[0];
#pragma unroll
    for (int i = 0; i < 32; ++i) s += pooled[(size_t)n * 32 + i] * W3[i];
    convbuf[(size_t)n * LATENT + 96] = tanhf(s / deg[n]);
}

// ----------------- fused conv(13x13,1->64) + bias + maxpool ----------------
// Implicit GEMM: M = 16 pixels (one row h, 16 consecutive w), N = 16 output
// channels per wave, K = 169 padded to 192. Block = 128 thr (4 waves = 4 oc
// tiles sharing one im2col tile). Each block owns one (graph, pool-row, wtile)
// and keeps the max over the 64 conv rows of the pool group in registers.
// grid = 8 * 30 * 7.

__global__ void conv_pool_wmma(const float* __restrict__ img,    // [NN][97]
                               const float* __restrict__ convW,  // [64][169]
                               const float* __restrict__ convB,  // [64]
                               float* __restrict__ out) {        // [8][64][30][97]
    __shared__ _Float16 Bw[OC][KPAD];      // weights, oc-major (pairs along K)
    __shared__ _Float16 At[16][KPAD];      // im2col tile, pixel-major

    const int tid  = threadIdx.x;          // 0..127
    const int lane = tid & 31;
    const int wave = tid >> 5;             // 0..3
    const int m    = lane & 15;
    const int half = lane >> 4;

    int blk = blockIdx.x;                  // b*30*7 + pr*7 + wt
    const int wt = blk % 7;   blk /= 7;
    const int pr = blk % KPOOL; blk /= KPOOL;
    const int b  = blk;
    const int w0  = wt * 16;
    const int ocb = wave * 16;

    // stage weights (f32 -> f16) once per block; zero-pad K to 192
    for (int i = tid; i < OC * KPAD; i += 128) {
        int oc = i / KPAD, k = i % KPAD;
        Bw[oc][k] = (k < KSZ) ? (_Float16)convW[oc * KSZ + k] : (_Float16)0.f;
    }

    v8f mx;
#pragma unroll
    for (int r = 0; r < 8; ++r) mx[r] = -3.0e38f;

    for (int hl = 0; hl < POOLH; ++hl) {
        const int h = pr * POOLH + hl;

        // im2col: 16 pixels x 192 K values (zero pad outside image / K>=169)
        for (int i = tid; i < 16 * KPAD; i += 128) {
            int pix = i / KPAD, k = i % KPAD;
            float v = 0.f;
            if (k < KSZ) {
                int kh = k / 13, kw = k % 13;
                int hh = h + kh - 6;
                int ww = w0 + pix + kw - 6;
                if (hh >= 0 && hh < NPER && ww >= 0 && ww < LATENT)
                    v = img[((size_t)b * NPER + hh) * LATENT + ww];
            }
            At[pix][k] = (_Float16)v;
        }
        __syncthreads();

        v8f c = {};
#pragma unroll
        for (int kk = 0; kk < KPAD / 32; ++kk) {
            const int base = kk * 32;
            v16h a, bf;
#pragma unroll
            for (int r = 0; r < 8; ++r) {
                // A 16x32 f16 layout: lanes0-15 K={0..7,16..23}, lanes16-31 +8
                const int ka = base + ((r < 4) ? 2 * r : 16 + 2 * (r - 4)) + half * 8;
                a[2 * r]     = At[m][ka];
                a[2 * r + 1] = At[m][ka + 1];
                // B 32x16 f16 layout: same K order as A: lanes0-15 K={0..7,16..23}, lanes16-31 +8
                const int kb = base + ((r < 4) ? 2 * r : 16 + 2 * (r - 4)) + half * 8;
                bf[2 * r]     = Bw[ocb + m][kb];
                bf[2 * r + 1] = Bw[ocb + m][kb + 1];
            }
            c = __builtin_amdgcn_wmma_f32_16x16x32_f16(false, a, false, bf,
                                                       (short)0, c, false, false);
        }
#pragma unroll
        for (int r = 0; r < 8; ++r) mx[r] = fmaxf(mx[r], c[r]);
        __syncthreads();
    }

    const int   oc = ocb + m;
    const float bb = convB[oc];
#pragma unroll
    for (int r = 0; r < 8; ++r) {
        const int pix = r + 8 * half;
        const int w   = w0 + pix;
        if (w < LATENT)
            out[(((size_t)b * OC + oc) * KPOOL + pr) * LATENT + w] = mx[r] + bb;
    }
}

// ------------------------------- launcher ----------------------------------

static inline int blks(long n, int t) { return (int)((n + t - 1) / t); }

extern "C" void kernel_launch(void* const* d_in, const int* in_sizes, int n_in,
                              void* d_out, int out_size, void* d_ws, size_t ws_size,
                              hipStream_t stream) {
    const float* nodeFeats = (const float*)d_in[0];
    const int*   src       = (const int*)d_in[1];
    const int*   dst       = (const int*)d_in[2];
    const float* W0 = (const float*)d_in[3];  const float* b0 = (const float*)d_in[4];
    const float* W1 = (const float*)d_in[5];  const float* b1 = (const float*)d_in[6];
    const float* W2 = (const float*)d_in[7];  const float* b2 = (const float*)d_in[8];
    const float* W3 = (const float*)d_in[9];  const float* b3 = (const float*)d_in[10];
    const float* convW = (const float*)d_in[11];
    const float* convB = (const float*)d_in[12];
    float* out = (float*)d_out;

    // workspace layout (~17.8 MB of f32)
    float* ws      = (float*)d_ws;
    float* deg     = ws;                         // NN
    float* pooled  = ws + 16384;                 // NN*FEAT (max)
    float* xa      = pooled + (size_t)NN * FEAT; // NN*32
    float* xb      = xa + (size_t)NN * 32;       // NN*32
    float* convbuf = xb + (size_t)NN * 32;       // NN*97

    // node degrees (+1 self term)
    fill_f32<<<blks(NN, 256), 256, 0, stream>>>(deg, 1.0f, NN);
    deg_scatter<<<blks(NEDGE, 256), 256, 0, stream>>>(dst, deg, NEDGE);

    // layer 0: 128 -> 32
    copy_f32<<<blks((long)NN * FEAT, 256), 256, 0, stream>>>(pooled, nodeFeats, NN * FEAT);
    edge_scatter<<<blks((long)NEDGE * FEAT, 256), 256, 0, stream>>>(nodeFeats, src, dst, pooled, NEDGE, 7);
    gnn_gemm32_wmma<<<NN / 16, 64, 0, stream>>>(pooled, FEAT, W0, b0, deg, xa, convbuf, 0);

    // layer 1: 32 -> 32
    copy_f32<<<blks((long)NN * 32, 256), 256, 0, stream>>>(pooled, xa, NN * 32);
    edge_scatter<<<blks((long)NEDGE * 32, 256), 256, 0, stream>>>(xa, src, dst, pooled, NEDGE, 5);
    gnn_gemm32_wmma<<<NN / 16, 64, 0, stream>>>(pooled, 32, W1, b1, deg, xb, convbuf, 32);

    // layer 2: 32 -> 32
    copy_f32<<<blks((long)NN * 32, 256), 256, 0, stream>>>(pooled, xb, NN * 32);
    edge_scatter<<<blks((long)NEDGE * 32, 256), 256, 0, stream>>>(xb, src, dst, pooled, NEDGE, 5);
    gnn_gemm32_wmma<<<NN / 16, 64, 0, stream>>>(pooled, 32, W2, b2, deg, xa, convbuf, 64);

    // layer 3: 32 -> 1
    copy_f32<<<blks((long)NN * 32, 256), 256, 0, stream>>>(pooled, xa, NN * 32);
    edge_scatter<<<blks((long)NEDGE * 32, 256), 256, 0, stream>>>(xa, src, dst, pooled, NEDGE, 5);
    gnn_layer3<<<blks(NN, 256), 256, 0, stream>>>(pooled, W3, b3, deg, convbuf);

    // fused conv + bias + maxpool -> [8,64,30,97]
    conv_pool_wmma<<<NB * KPOOL * 7, 128, 0, stream>>>(convbuf, convW, convB, out);
}
